// Detect_21466246545870
// MI455X (gfx1250) — matrix operand order repaired and checked by reading, not compiled, not run on hardware
//
#include <hip/hip_runtime.h>
#include <cstdint>

// ---------------- CDNA5 WMMA types ----------------
typedef __attribute__((ext_vector_type(16))) __bf16 bf16x16;
typedef __attribute__((ext_vector_type(8)))  float  f32x8;

union Frag {
    bf16x16 v;
    uint4 q[2];
    unsigned short s[16];
};

__device__ __forceinline__ unsigned short f2bf(float f) {
    unsigned int u = __float_as_uint(f);
    u += 0x7FFFu + ((u >> 16) & 1u);   // round-to-nearest-even
    return (unsigned short)(u >> 16);
}

__device__ __forceinline__ uint4 ld4(const unsigned short* p) {
    return *reinterpret_cast<const uint4*>(p);
}

__device__ __forceinline__ float silu(float v) {
    return v / (1.f + __expf(-v));
}

// ---------------- layout packing ----------------
__global__ void cvt_bf16_kernel(const float* __restrict__ src, unsigned short* __restrict__ dst, int n) {
    int i = blockIdx.x * blockDim.x + threadIdx.x;
    if (i < n) dst[i] = f2bf(src[i]);
}

// f32 NCHW [B][C][HW] -> bf16 channels-last [B*HW][C]; i indexes src (coalesced reads)
__global__ void nchw_to_cl_bf16_kernel(const float* __restrict__ src, unsigned short* __restrict__ dst,
                                       int C, int HW, int total) {
    int i = blockIdx.x * blockDim.x + threadIdx.x;
    if (i >= total) return;
    int pos = i % HW; int rest = i / HW; int c = rest % C; int b = rest / C;
    dst[((size_t)b * HW + pos) * C + c] = f2bf(src[i]);
}

// OIHW [C][C][3][3] -> [C][9][C] bf16
__global__ void pack3x3_kernel(const float* __restrict__ src, unsigned short* __restrict__ dst, int C, int n) {
    int i = blockIdx.x * blockDim.x + threadIdx.x;
    if (i >= n) return;
    int ci = i % C; int rest = i / C; int t = rest % 9; int co = rest / 9;
    dst[i] = f2bf(src[((size_t)co * C + ci) * 9 + t]);
}

// rows 0..3 = reg_pred_w, row 4 = obj_pred_w, rows 5..15 = 0  -> [16][C] bf16
__global__ void pack_ro_kernel(const float* __restrict__ regw, const float* __restrict__ objw,
                               unsigned short* __restrict__ dst, int C) {
    int i = blockIdx.x * blockDim.x + threadIdx.x;
    if (i >= 16 * C) return;
    int ci = i % C; int row = i / C;
    float v = (row < 4) ? regw[(size_t)row * C + ci] : ((row == 4) ? objw[ci] : 0.f);
    dst[i] = f2bf(v);
}

// ---------------- fragment load helpers (for software pipelining) ----------------
__device__ __forceinline__ void load_step_1x1(const unsigned short* __restrict__ arow,
                                              const unsigned short* __restrict__ wpk,
                                              int kc, int C, int nbase, int ln, int kb, int kbq,
                                              Frag& af, Frag* bf) {
    af.q[0] = ld4(arow + kc + kb);
    af.q[1] = ld4(arow + kc + kb + 16);
#pragma unroll
    for (int t = 0; t < 4; ++t) {
        const unsigned short* br = wpk + (size_t)(nbase + t * 16 + ln) * C + kc + kb;
        bf[t].q[0] = ld4(br);
        bf[t].q[1] = ld4(br + 16);
    }
}

__device__ __forceinline__ void load_step_3x3(const unsigned short* __restrict__ act,
                                              const unsigned short* __restrict__ wpk,
                                              int s, int kcsh, int KCm1,
                                              int h, int w, int H, int W, int HW, int b, int C,
                                              int nbase, int ln, int kb, int kbq,
                                              Frag& af, Frag* bf) {
    int tap = s >> kcsh;
    int kc = (s & KCm1) << 5;
    int hy = h + tap / 3 - 1;
    int wx = w + tap % 3 - 1;
    bool ok = (hy >= 0) & (hy < H) & (wx >= 0) & (wx < W);
    const unsigned short* ar = act + ((size_t)b * HW + (size_t)hy * W + wx) * C + kc;
    if (ok) {
        af.q[0] = ld4(ar + kb);
        af.q[1] = ld4(ar + kb + 16);
    } else {
        af.q[0] = make_uint4(0u, 0u, 0u, 0u);
        af.q[1] = make_uint4(0u, 0u, 0u, 0u);
    }
#pragma unroll
    for (int t = 0; t < 4; ++t) {
        const unsigned short* br = wpk + (((size_t)(nbase + t * 16 + ln) * 9 + tap) * C + kc + kb);
        bf[t].q[0] = ld4(br);
        bf[t].q[1] = ld4(br + 16);
    }
}

// ---------------- 1x1 conv + SiLU, bf16 channels-last in/out (stem) ----------------
__global__ __launch_bounds__(256) void conv1x1_silu_kernel(
    const unsigned short* __restrict__ act, // [B*HW][C] bf16
    const unsigned short* __restrict__ wpk, // [C][C] bf16
    const float* __restrict__ bias,         // [C]
    unsigned short* __restrict__ out,       // [B*HW][C] bf16
    int C, int Mtiles, int Ngroups)
{
    int gw = (blockIdx.x * blockDim.x + threadIdx.x) >> 5;
    if (gw >= Mtiles * Ngroups) return;
    int lane = threadIdx.x & 31;
    int mtile = gw / Ngroups;
    int ng = gw % Ngroups;
    int nbase = ng * 64;
    int ln = lane & 15;
    int half = lane >> 4;
    int kb = half * 8;     // A-fragment K base within 32-chunk
    int kbq = half * 16;   // unused; B fragments also use kb

    const unsigned short* arow = act + (size_t)(mtile * 16 + ln) * C;

    f32x8 acc[4] = {};
    Frag ac, bc[4], an, bn[4];
    load_step_1x1(arow, wpk, 0, C, nbase, ln, kb, kbq, ac, bc);
    for (int kc = 32; kc < C; kc += 32) {
        load_step_1x1(arow, wpk, kc, C, nbase, ln, kb, kbq, an, bn);
#pragma unroll
        for (int t = 0; t < 4; ++t)
            acc[t] = __builtin_amdgcn_wmma_f32_16x16x32_bf16(false, ac.v, false, bc[t].v,
                                                             (short)0, acc[t], false, false);
        ac = an;
#pragma unroll
        for (int t = 0; t < 4; ++t) bc[t] = bn[t];
    }
#pragma unroll
    for (int t = 0; t < 4; ++t)
        acc[t] = __builtin_amdgcn_wmma_f32_16x16x32_bf16(false, ac.v, false, bc[t].v,
                                                         (short)0, acc[t], false, false);

#pragma unroll
    for (int t = 0; t < 4; ++t) {
        int co = nbase + t * 16 + ln;
        float bi = bias[co];
#pragma unroll
        for (int r = 0; r < 8; ++r) {
            int m = mtile * 16 + half * 8 + r;
            out[(size_t)m * C + co] = f2bf(silu(acc[t][r] + bi));
        }
    }
}

// ---------------- 3x3 SAME conv + SiLU, bf16 channels-last in/out ----------------
__global__ __launch_bounds__(256) void conv3x3_silu_kernel(
    const unsigned short* __restrict__ act, // [B*HW][C] bf16
    const unsigned short* __restrict__ wpk, // [C][9][C] bf16
    const float* __restrict__ bias,
    unsigned short* __restrict__ out,       // [B*HW][C] bf16
    int C, int H, int W, int Mtiles, int Ngroups)
{
    int gw = (blockIdx.x * blockDim.x + threadIdx.x) >> 5;
    if (gw >= Mtiles * Ngroups) return;
    int lane = threadIdx.x & 31;
    int mtile = gw / Ngroups;
    int ng = gw % Ngroups;
    int nbase = ng * 64;
    int ln = lane & 15;
    int half = lane >> 4;
    int kb = half * 8;
    int kbq = half * 16;
    int HW = H * W;

    int ma = mtile * 16 + ln;
    int b = ma / HW, pos = ma % HW;
    int h = pos / W, w = pos % W;

    const int KC = C >> 5;               // K-chunks per tap (4/8/16, power of two)
    const int kcsh = __builtin_ctz(KC);
    const int KCm1 = KC - 1;
    const int steps = 9 << kcsh;

    f32x8 acc[4] = {};
    Frag ac, bc[4], an, bn[4];
    load_step_3x3(act, wpk, 0, kcsh, KCm1, h, w, H, W, HW, b, C, nbase, ln, kb, kbq, ac, bc);
    for (int s = 0; s < steps - 1; ++s) {
        load_step_3x3(act, wpk, s + 1, kcsh, KCm1, h, w, H, W, HW, b, C, nbase, ln, kb, kbq, an, bn);
#pragma unroll
        for (int t = 0; t < 4; ++t)
            acc[t] = __builtin_amdgcn_wmma_f32_16x16x32_bf16(false, ac.v, false, bc[t].v,
                                                             (short)0, acc[t], false, false);
        ac = an;
#pragma unroll
        for (int t = 0; t < 4; ++t) bc[t] = bn[t];
    }
#pragma unroll
    for (int t = 0; t < 4; ++t)
        acc[t] = __builtin_amdgcn_wmma_f32_16x16x32_bf16(false, ac.v, false, bc[t].v,
                                                         (short)0, acc[t], false, false);

#pragma unroll
    for (int t = 0; t < 4; ++t) {
        int co = nbase + t * 16 + ln;
        float bi = bias[co];
#pragma unroll
        for (int r = 0; r < 8; ++r) {
            int m = mtile * 16 + half * 8 + r;
            out[(size_t)m * C + co] = f2bf(silu(acc[t][r] + bi));
        }
    }
}

// ---------------- prediction heads (cls 5x16 tiles + reg/obj 1x16 tile) ----------------
__global__ __launch_bounds__(256) void head_kernel(
    const unsigned short* __restrict__ clsf, // [M][C] bf16
    const unsigned short* __restrict__ regf, // [M][C] bf16
    const unsigned short* __restrict__ clsw, // [80][C] bf16
    const unsigned short* __restrict__ wro,  // [16][C] bf16
    const float* __restrict__ clsb, const float* __restrict__ regb, const float* __restrict__ objb,
    float* __restrict__ raw,                 // [M][96] f32
    int C, int Mtiles)
{
    int gw = (blockIdx.x * blockDim.x + threadIdx.x) >> 5;
    if (gw >= Mtiles * 6) return;
    int lane = threadIdx.x & 31;
    int mtile = gw / 6;
    int nt = gw % 6;
    int ln = lane & 15;
    int half = lane >> 4;
    int kb = half * 8;
    int kbq = half * 16;

    const unsigned short* feat = (nt < 5) ? clsf : regf;
    const unsigned short* wbase = (nt < 5) ? (clsw + (size_t)(nt * 16) * C) : wro;
    const unsigned short* arow = feat + (size_t)(mtile * 16 + ln) * C;
    const unsigned short* brow0 = wbase + (size_t)ln * C;

    f32x8 acc = {};
    Frag ac, bc, an, bn;
    ac.q[0] = ld4(arow + kb);       ac.q[1] = ld4(arow + kb + 16);
    bc.q[0] = ld4(brow0 + kb);      bc.q[1] = ld4(brow0 + kb + 16);
    for (int kc = 32; kc < C; kc += 32) {
        an.q[0] = ld4(arow + kc + kb);       an.q[1] = ld4(arow + kc + kb + 16);
        bn.q[0] = ld4(brow0 + kc + kb);      bn.q[1] = ld4(brow0 + kc + kb + 16);
        acc = __builtin_amdgcn_wmma_f32_16x16x32_bf16(false, ac.v, false, bc.v,
                                                      (short)0, acc, false, false);
        ac = an; bc = bn;
    }
    acc = __builtin_amdgcn_wmma_f32_16x16x32_bf16(false, ac.v, false, bc.v,
                                                  (short)0, acc, false, false);

    int col = nt * 16 + ln;
    float bi = (nt < 5) ? clsb[col] : ((ln < 4) ? regb[ln] : ((ln == 4) ? objb[0] : 0.f));
#pragma unroll
    for (int r = 0; r < 8; ++r) {
        int m = mtile * 16 + half * 8 + r;
        raw[(size_t)m * 96 + col] = acc[r] + bi;
    }
}

// ---------------- decode: grid offset / exp / sigmoid / best class ----------------
__global__ void decode_kernel(const float* __restrict__ raw, float* __restrict__ det,
                              int HW, int W, float stride, int aoff, int total)
{
    int i = blockIdx.x * blockDim.x + threadIdx.x;
    if (i >= total) return;
    int b = i / HW, pos = i % HW;
    int h = pos / W, w = pos % W;
    const float* r = raw + (size_t)i * 96;
    float cx = (r[80] + (float)w) * stride;
    float cy = (r[81] + (float)h) * stride;
    float bw = __expf(r[82]) * stride;
    float bh = __expf(r[83]) * stride;
    float so = 1.f / (1.f + __expf(-r[84]));
    float best = -1.f; int lab = 0;
    for (int c = 0; c < 80; ++c) {
        float sc = so / (1.f + __expf(-r[c]));   // sigmoid(obj)*sigmoid(cls)
        if (sc > best) { best = sc; lab = c; }
    }
    float sm = (best > 0.15f) ? best : -1.f;
    float* d = det + ((size_t)b * 8400 + aoff + pos) * 8;
    d[0] = cx; d[1] = cy; d[2] = bw; d[3] = bh; d[4] = sm; d[5] = (float)lab;
}

// ---------------- NMS: one workgroup per image ----------------
__device__ __forceinline__ float iou_f(const float* a, const float* b) {
    float tlx = fmaxf(a[0], b[0]), tly = fmaxf(a[1], b[1]);
    float brx = fminf(a[2], b[2]), bry = fminf(a[3], b[3]);
    float inter = fmaxf(brx - tlx, 0.f) * fmaxf(bry - tly, 0.f);
    float aa = (a[2] - a[0]) * (a[3] - a[1]);
    float ab = (b[2] - b[0]) * (b[3] - b[1]);
    return inter / (aa + ab - inter + 1e-7f);
}

__global__ __launch_bounds__(1024) void nms_kernel(const float* __restrict__ det, float* __restrict__ out)
{
    __shared__ float s_sc[8400];
    __shared__ float s_bx[1000 * 4];   // class-offset xyxy
    __shared__ float s_ts[1000];
    __shared__ int   s_lab[1000];
    __shared__ int   s_keep[1000];
    int b = blockIdx.x, tid = threadIdx.x, bs = blockDim.x;
    const float* db = det + (size_t)b * 8400 * 8;
    float* ob = out + (size_t)b * 300 * 6;

    for (int i = tid; i < 8400; i += bs) s_sc[i] = db[(size_t)i * 8 + 4];
    for (int i = tid; i < 300 * 6; i += bs) ob[i] = 0.f;
    __syncthreads();

    // exact top-1000 via rank counting (matches lax.top_k stable tie-break)
    for (int i = tid; i < 8400; i += bs) {
        float si = s_sc[i];
        int rank = 0;
        for (int j = 0; j < 8400; ++j) {
            float sj = s_sc[j];
            rank += (sj > si) || ((sj == si) && (j < i));
        }
        if (rank < 1000) {
            const float* d = db + (size_t)i * 8;
            float off = d[5] * 4096.f;  // class-aware offset trick
            float hw2 = d[2] * 0.5f, hh2 = d[3] * 0.5f;
            s_bx[rank * 4 + 0] = d[0] - hw2 + off;
            s_bx[rank * 4 + 1] = d[1] - hh2 + off;
            s_bx[rank * 4 + 2] = d[0] + hw2 + off;
            s_bx[rank * 4 + 3] = d[1] + hh2 + off;
            s_ts[rank] = si;
            s_lab[rank] = (int)d[5];
            s_keep[rank] = 1;
        }
    }
    __syncthreads();

    // greedy suppression
    for (int i = 0; i < 1000; ++i) {
        if (s_keep[i]) {
            float ai[4] = { s_bx[i * 4 + 0], s_bx[i * 4 + 1], s_bx[i * 4 + 2], s_bx[i * 4 + 3] };
            for (int j = i + 1 + tid; j < 1000; j += bs) {
                if (s_keep[j] && iou_f(ai, &s_bx[j * 4]) > 0.45f) s_keep[j] = 0;
            }
        }
        __syncthreads();
    }

    if (tid == 0) {
        int cnt = 0;
        for (int i = 0; i < 1000 && cnt < 300; ++i) {
            if (s_keep[i] && s_ts[i] > 0.15f) {
                float off = (float)s_lab[i] * 4096.f;
                float* o = ob + cnt * 6;
                o[0] = s_bx[i * 4 + 0] - off;
                o[1] = s_bx[i * 4 + 1] - off;
                o[2] = s_bx[i * 4 + 2] - off;
                o[3] = s_bx[i * 4 + 3] - off;
                o[4] = (float)(s_lab[i] + 1);
                o[5] = s_ts[i];
                ++cnt;
            }
        }
    }
}

// ---------------- host launch ----------------
extern "C" void kernel_launch(void* const* d_in, const int* in_sizes, int n_in,
                              void* d_out, int out_size, void* d_ws, size_t ws_size,
                              hipStream_t stream)
{
    (void)in_sizes; (void)n_in; (void)out_size; (void)ws_size;
    const int CHs[3] = {128, 256, 512};
    const int Hs[3]  = {80, 40, 20};
    const int B = 8;
    const float strides[3] = {8.f, 16.f, 32.f};
    const int aoffs[3] = {0, 6400, 8000};

    const float* X[3] = {(const float*)d_in[0], (const float*)d_in[1], (const float*)d_in[2]};
    const float *stem_w[3], *stem_b[3], *clsc_w[3], *clsc_b[3], *regc_w[3], *regc_b[3],
                *clsp_w[3], *clsp_b[3], *regp_w[3], *regp_b[3], *objp_w[3], *objp_b[3];
    for (int s = 0; s < 3; ++s) {
        int base = 3 + s * 12;
        stem_w[s] = (const float*)d_in[base + 0];  stem_b[s] = (const float*)d_in[base + 1];
        clsc_w[s] = (const float*)d_in[base + 2];  clsc_b[s] = (const float*)d_in[base + 3];
        regc_w[s] = (const float*)d_in[base + 4];  regc_b[s] = (const float*)d_in[base + 5];
        clsp_w[s] = (const float*)d_in[base + 6];  clsp_b[s] = (const float*)d_in[base + 7];
        regp_w[s] = (const float*)d_in[base + 8];  regp_b[s] = (const float*)d_in[base + 9];
        objp_w[s] = (const float*)d_in[base + 10]; objp_b[s] = (const float*)d_in[base + 11];
    }

    // carve workspace
    char* p = (char*)d_ws;
    auto alloc = [&](size_t bytes) -> char* {
        char* r = p;
        p += (bytes + 255) & ~(size_t)255;
        return r;
    };
    unsigned short *wstem[3], *wcls3[3], *wreg3[3], *wclsp[3], *wro[3];
    unsigned short *xbf[3], *fact[3], *cact[3], *ract[3];
    float* raw[3];
    for (int s = 0; s < 3; ++s) {
        size_t C = CHs[s], HW = (size_t)Hs[s] * Hs[s];
        wstem[s] = (unsigned short*)alloc(C * C * 2);
        wcls3[s] = (unsigned short*)alloc(C * C * 9 * 2);
        wreg3[s] = (unsigned short*)alloc(C * C * 9 * 2);
        wclsp[s] = (unsigned short*)alloc(80 * C * 2);
        wro[s]   = (unsigned short*)alloc(16 * C * 2);
        xbf[s]   = (unsigned short*)alloc((size_t)B * HW * C * 2);
        fact[s]  = (unsigned short*)alloc((size_t)B * HW * C * 2);
        cact[s]  = (unsigned short*)alloc((size_t)B * HW * C * 2);
        ract[s]  = (unsigned short*)alloc((size_t)B * HW * C * 2);
        raw[s]   = (float*)alloc((size_t)B * HW * 96 * 4);
    }
    float* det = (float*)alloc((size_t)B * 8400 * 8 * 4);

    // pack weights + convert inputs to bf16 channels-last
    for (int s = 0; s < 3; ++s) {
        int C = CHs[s], HW = Hs[s] * Hs[s];
        int n1 = C * C;
        cvt_bf16_kernel<<<(n1 + 255) / 256, 256, 0, stream>>>(stem_w[s], wstem[s], n1);
        int n9 = C * C * 9;
        pack3x3_kernel<<<(n9 + 255) / 256, 256, 0, stream>>>(clsc_w[s], wcls3[s], C, n9);
        pack3x3_kernel<<<(n9 + 255) / 256, 256, 0, stream>>>(regc_w[s], wreg3[s], C, n9);
        int nc = 80 * C;
        cvt_bf16_kernel<<<(nc + 255) / 256, 256, 0, stream>>>(clsp_w[s], wclsp[s], nc);
        pack_ro_kernel<<<(16 * C + 255) / 256, 256, 0, stream>>>(regp_w[s], objp_w[s], wro[s], C);
        int nx = B * C * HW;
        nchw_to_cl_bf16_kernel<<<(nx + 255) / 256, 256, 0, stream>>>(X[s], xbf[s], C, HW, nx);
    }

    // conv pipeline per scale
    for (int s = 0; s < 3; ++s) {
        int C = CHs[s], H = Hs[s], W = Hs[s], HW = H * W, M = B * HW;
        int Mt = M / 16, Ng = C / 64;
        int waves = Mt * Ng;
        int blocks = (waves + 7) / 8;   // 256 threads = 8 waves
        conv1x1_silu_kernel<<<blocks, 256, 0, stream>>>(xbf[s], wstem[s], stem_b[s], fact[s], C, Mt, Ng);
        conv3x3_silu_kernel<<<blocks, 256, 0, stream>>>(fact[s], wcls3[s], clsc_b[s], cact[s], C, H, W, Mt, Ng);
        conv3x3_silu_kernel<<<blocks, 256, 0, stream>>>(fact[s], wreg3[s], regc_b[s], ract[s], C, H, W, Mt, Ng);
        int wavesH = Mt * 6;
        int blocksH = (wavesH + 7) / 8;
        head_kernel<<<blocksH, 256, 0, stream>>>(cact[s], ract[s], wclsp[s], wro[s],
                                                 clsp_b[s], regp_b[s], objp_b[s], raw[s], C, Mt);
        decode_kernel<<<(M + 255) / 256, 256, 0, stream>>>(raw[s], det, HW, W, strides[s], aoffs[s], M);
    }

    nms_kernel<<<B, 1024, 0, stream>>>(det, (float*)d_out);
}
